// L2Attention_21285857919719
// MI455X (gfx1250) — matrix operand order repaired and checked by reading, not compiled, not run on hardware
//
#include <hip/hip_runtime.h>

typedef __bf16 bf16;
typedef bf16 v16bf __attribute__((ext_vector_type(16)));
typedef bf16 v8bf  __attribute__((ext_vector_type(8)));
typedef float v8f  __attribute__((ext_vector_type(8)));

#define BATCH 4
#define SEQ   1024
#define DIM   1024
#define HEADS 16
#define HD    64              // head dim
#define TOK   (BATCH*SEQ)     // 4096

// ---------------- helpers ----------------

static __device__ inline v16bf comb(v8bf lo, v8bf hi) {
  return __builtin_shufflevector(lo, hi, 0,1,2,3,4,5,6,7,8,9,10,11,12,13,14,15);
}

static __device__ inline v8bf cvt8(const float* __restrict__ p) {
  v8bf r;
#pragma unroll
  for (int i = 0; i < 8; ++i) r[i] = (bf16)p[i];
  return r;
}

// A operand: 16(M) x 32(K). lane&15 = row M; lanes 0-15 hold K=k..k+15,
// lanes 16-31 hold K=k+16..k+31 (matches loadB_bf).
static __device__ inline v16bf loadA_bf(const bf16* __restrict__ src, int ld,
                                        int row_base, int k_base, int lane) {
  int row = row_base + (lane & 15);
  int k0  = k_base + ((lane >> 4) << 4);
  const bf16* p = src + (size_t)row * ld + k0;
  return comb(*(const v8bf*)p, *(const v8bf*)(p + 8));
}
// B operand: 32(K) x 16(N). src stored as rows of B^T, i.e. src[col][k].
// lane&15 = col N; lanes 0-15 hold K=k..k+15, lanes 16-31 hold K=k+16..k+31.
static __device__ inline v16bf loadB_bf(const bf16* __restrict__ src, int ld,
                                        int col_base, int k_base, int lane) {
  int col = col_base + (lane & 15);
  int k0  = k_base + ((lane >> 4) << 4);
  const bf16* p = src + (size_t)col * ld + k0;
  return comb(*(const v8bf*)p, *(const v8bf*)(p + 8));
}

static __device__ inline v8f wmma_bf(v16bf a, v16bf b, v8f c) {
  return __builtin_amdgcn_wmma_f32_16x16x32_bf16(false, a, false, b,
                                                 (short)0, c, false, false);
}

// ---------------- kernel 0: fp32 -> bf16 pre-conversion ----------------
// Memory bound (~46 MB total => ~2us at 23.3 TB/s); removes all cvt VALU and
// halves the streamed bytes in the compute-bound WMMA loops.
__global__ __launch_bounds__(256) void k_cvt_bf16(const float* __restrict__ src,
                                                  bf16* __restrict__ dst, int n) {
  int stride = gridDim.x * blockDim.x;
  for (int i = blockIdx.x * blockDim.x + threadIdx.x; i * 8 < n; i += stride) {
    *(v8bf*)(dst + (size_t)i * 8) = cvt8(src + (size_t)i * 8);
  }
}

// ---------------- kernel 1: Q projection + ||q||^2 ----------------
// C[m=feature, n=token] = sum_k Wq[f,k] * x[t,k]; wave computes 64x64 tile.
// Feature blocks (64) align exactly with one head (d=64), so the wave also
// produces sq for its 64 tokens with one cross-half shuffle.
__global__ __launch_bounds__(256) void k_proj_q(
    const bf16* __restrict__ x, const bf16* __restrict__ Wq,
    const float* __restrict__ bq, bf16* __restrict__ q_s,
    float* __restrict__ sq_s) {
  int lane = threadIdx.x & 31, hi = lane >> 4;
  int wave = blockIdx.x * 8 + (threadIdx.x >> 5);
  int fm = (wave & 15) * 64;   // feature base (one head)
  int tn = (wave >> 4) * 64;   // token base
  v8f acc[4][4];
#pragma unroll
  for (int i = 0; i < 4; ++i)
#pragma unroll
    for (int j = 0; j < 4; ++j)
#pragma unroll
      for (int g = 0; g < 8; ++g) acc[i][j][g] = 0.f;

  for (int kk = 0; kk < DIM; kk += 32) {
    v16bf a[4], b[4];
#pragma unroll
    for (int i = 0; i < 4; ++i) a[i] = loadA_bf(Wq, DIM, fm + 16 * i, kk, lane);
#pragma unroll
    for (int j = 0; j < 4; ++j) b[j] = loadB_bf(x, DIM, tn + 16 * j, kk, lane);
#pragma unroll
    for (int i = 0; i < 4; ++i)
#pragma unroll
      for (int j = 0; j < 4; ++j) acc[i][j] = wmma_bf(a[i], b[j], acc[i][j]);
  }

  float sqa[4] = {0.f, 0.f, 0.f, 0.f};
#pragma unroll
  for (int i = 0; i < 4; ++i) {
    int f0 = fm + 16 * i + hi * 8;
#pragma unroll
    for (int j = 0; j < 4; ++j) {
      int tok = tn + 16 * j + (lane & 15);
      v8bf ov;
#pragma unroll
      for (int g = 0; g < 8; ++g) {
        float y = acc[i][j][g] + bq[f0 + g];
        sqa[j] += y * y;
        ov[g] = (bf16)y;
      }
      int b_ = tok >> 10, seq = tok & 1023;
      int h_ = f0 >> 6, d0 = f0 & 63;
      bf16* dst = q_s + (((size_t)(b_ * HEADS + h_) * SEQ + seq) * HD + d0);
      *(v8bf*)dst = ov;
    }
  }
  int h_ = fm >> 6;
#pragma unroll
  for (int j = 0; j < 4; ++j) {
    float s = sqa[j] + __shfl_xor(sqa[j], 16, 32);
    if (hi == 0) {
      int tok = tn + 16 * j + lane;
      sq_s[(size_t)((tok >> 10) * HEADS + h_) * SEQ + (tok & 1023)] = s;
    }
  }
}

// ---------------- kernel 2: V projection -> V^T scratch ----------------
// C[m=token, n=feature] = sum_k x[t,k] * Wv[f,k]; writes vT[bh][d][seq].
__global__ __launch_bounds__(256) void k_proj_v(
    const bf16* __restrict__ x, const bf16* __restrict__ Wv,
    const float* __restrict__ bv, bf16* __restrict__ vT_s) {
  int lane = threadIdx.x & 31, hi = lane >> 4;
  int wave = blockIdx.x * 8 + (threadIdx.x >> 5);
  int fn = (wave & 15) * 64;   // feature base
  int tm = (wave >> 4) * 64;   // token base
  v8f acc[4][4];
#pragma unroll
  for (int i = 0; i < 4; ++i)
#pragma unroll
    for (int j = 0; j < 4; ++j)
#pragma unroll
      for (int g = 0; g < 8; ++g) acc[i][j][g] = 0.f;

  for (int kk = 0; kk < DIM; kk += 32) {
    v16bf a[4], b[4];
#pragma unroll
    for (int i = 0; i < 4; ++i) a[i] = loadA_bf(x, DIM, tm + 16 * i, kk, lane);
#pragma unroll
    for (int j = 0; j < 4; ++j) b[j] = loadB_bf(Wv, DIM, fn + 16 * j, kk, lane);
#pragma unroll
    for (int i = 0; i < 4; ++i)
#pragma unroll
      for (int j = 0; j < 4; ++j) acc[i][j] = wmma_bf(a[i], b[j], acc[i][j]);
  }
#pragma unroll
  for (int j = 0; j < 4; ++j) {
    int f = fn + 16 * j + (lane & 15);
    float bias = bv[f];
    int h_ = f >> 6, d_ = f & 63;
#pragma unroll
    for (int i = 0; i < 4; ++i) {
      int tok0 = tm + 16 * i + hi * 8;   // 8 consecutive tokens
      v8bf ov;
#pragma unroll
      for (int g = 0; g < 8; ++g) ov[g] = (bf16)(acc[i][j][g] + bias);
      int b_ = tok0 >> 10, seq0 = tok0 & 1023;
      bf16* dst = vT_s + (((size_t)(b_ * HEADS + h_) * HD + d_) * SEQ + seq0);
      *(v8bf*)dst = ov;
    }
  }
}

// ---------------- kernel 3: flash L2-attention (transposed) ----------------
// Per wave: one (bh, 16-query block). S^T = Q_keys * Q_queries^T so the
// softmax reduction over keys is lane-local (+1 shfl to merge lane halves).
// out^T = V^T * P^T accumulated in 4 WMMA tiles; P^T re-shaped via a 1KB
// per-wave LDS slice (same-wave ds ops are in-order; no barrier needed).
__global__ __launch_bounds__(256) void k_attn(
    const bf16* __restrict__ q_s, const bf16* __restrict__ vT_s,
    const float* __restrict__ sq_s, bf16* __restrict__ outflat) {
  __shared__ __align__(16) bf16 plds[8][16][32];
  int lane = threadIdx.x & 31, hi = lane >> 4;
  int wv = threadIdx.x >> 5;
  int wave = blockIdx.x * 8 + wv;
  int bh = wave >> 6;
  int qb = (wave & 63) * 16;
  const bf16* Q  = q_s  + (size_t)bh * SEQ * HD;
  const bf16* VT = vT_s + (size_t)bh * HD * SEQ;
  const float* SQ = sq_s + (size_t)bh * SEQ;
  const float scale = 0.125f;  // d^-0.5

  v16bf bq0 = loadB_bf(Q, HD, qb, 0, lane);
  v16bf bq1 = loadB_bf(Q, HD, qb, 32, lane);

  v8f o[4];
#pragma unroll
  for (int r = 0; r < 4; ++r)
#pragma unroll
    for (int g = 0; g < 8; ++g) o[r][g] = 0.f;
  float m = -1e30f, l = 0.f;

  for (int jb = 0; jb < SEQ; jb += 32) {
    v8f s0, s1;
#pragma unroll
    for (int g = 0; g < 8; ++g) { s0[g] = 0.f; s1[g] = 0.f; }
    s0 = wmma_bf(loadA_bf(Q, HD, jb, 0, lane), bq0, s0);
    s0 = wmma_bf(loadA_bf(Q, HD, jb, 32, lane), bq1, s0);
    s1 = wmma_bf(loadA_bf(Q, HD, jb + 16, 0, lane), bq0, s1);
    s1 = wmma_bf(loadA_bf(Q, HD, jb + 16, 32, lane), bq1, s1);

    const float* sqp = SQ + jb + hi * 8;
    float z0[8], z1[8], tmax = -1e30f;
#pragma unroll
    for (int g = 0; g < 8; ++g) {
      z0[g] = (2.f * s0[g] - sqp[g]) * scale;
      z1[g] = (2.f * s1[g] - sqp[16 + g]) * scale;
      tmax = fmaxf(tmax, fmaxf(z0[g], z1[g]));
    }
    tmax = fmaxf(tmax, __shfl_xor(tmax, 16, 32));
    float mn = fmaxf(m, tmax);
    float alpha = __expf(m - mn);
    float lloc = 0.f;
    v8bf p0, p1;
#pragma unroll
    for (int g = 0; g < 8; ++g) {
      float pa = __expf(z0[g] - mn);
      float pb = __expf(z1[g] - mn);
      lloc += pa + pb;
      p0[g] = (bf16)pa;
      p1[g] = (bf16)pb;
    }
    lloc += __shfl_xor(lloc, 16, 32);
    l = l * alpha + lloc;
    m = mn;
#pragma unroll
    for (int r = 0; r < 4; ++r)
#pragma unroll
      for (int g = 0; g < 8; ++g) o[r][g] *= alpha;

    int n = lane & 15;
    *(v8bf*)&plds[wv][n][hi * 8]      = p0;   // keys jb+8hi..+7
    *(v8bf*)&plds[wv][n][16 + hi * 8] = p1;   // keys jb+16+8hi..+7
    v16bf bp = comb(*(const v8bf*)&plds[wv][n][hi * 16],
                    *(const v8bf*)&plds[wv][n][hi * 16 + 8]);
#pragma unroll
    for (int r = 0; r < 4; ++r)
      o[r] = wmma_bf(loadA_bf(VT, SEQ, r * 16, jb, lane), bp, o[r]);
  }

  float inv = 1.f / l;
  int b_ = bh >> 4, h_ = bh & 15;
  int seq = qb + (lane & 15);
  bf16* dst = outflat + (size_t)(b_ * SEQ + seq) * DIM + h_ * HD;
#pragma unroll
  for (int r = 0; r < 4; ++r) {
    v8bf ov;
#pragma unroll
    for (int g = 0; g < 8; ++g) ov[g] = (bf16)(o[r][g] * inv);
    *(v8bf*)(dst + r * 16 + hi * 8) = ov;
  }
}

// ---------------- kernel 4: output projection ----------------
// C[m=feature, n=token] = sum_k Wo[f,k] * outflat[t,k] + bo[f], fp32 out.
__global__ __launch_bounds__(256) void k_proj_o(
    const bf16* __restrict__ of, const bf16* __restrict__ Wo,
    const float* __restrict__ bo, float* __restrict__ out) {
  int lane = threadIdx.x & 31, hi = lane >> 4;
  int wave = blockIdx.x * 8 + (threadIdx.x >> 5);
  int fm = (wave & 15) * 64;
  int tn = (wave >> 4) * 64;
  v8f acc[4][4];
#pragma unroll
  for (int i = 0; i < 4; ++i)
#pragma unroll
    for (int j = 0; j < 4; ++j)
#pragma unroll
      for (int g = 0; g < 8; ++g) acc[i][j][g] = 0.f;

  for (int kk = 0; kk < DIM; kk += 32) {
    v16bf a[4], b[4];
#pragma unroll
    for (int i = 0; i < 4; ++i) a[i] = loadA_bf(Wo, DIM, fm + 16 * i, kk, lane);
#pragma unroll
    for (int j = 0; j < 4; ++j) b[j] = loadB_bf(of, DIM, tn + 16 * j, kk, lane);
#pragma unroll
    for (int i = 0; i < 4; ++i)
#pragma unroll
      for (int j = 0; j < 4; ++j) acc[i][j] = wmma_bf(a[i], b[j], acc[i][j]);
  }
#pragma unroll
  for (int i = 0; i < 4; ++i) {
    int f0 = fm + 16 * i + hi * 8;
#pragma unroll
    for (int j = 0; j < 4; ++j) {
      int tok = tn + 16 * j + (lane & 15);
      v8f y;
#pragma unroll
      for (int g = 0; g < 8; ++g) y[g] = acc[i][j][g] + bo[f0 + g];
      *(v8f*)(out + (size_t)tok * DIM + f0) = y;
    }
  }
}

// ---------------- launcher ----------------
extern "C" void kernel_launch(void* const* d_in, const int* in_sizes, int n_in,
                              void* d_out, int out_size, void* d_ws, size_t ws_size,
                              hipStream_t stream) {
  const float* x  = (const float*)d_in[0];
  const float* Wq = (const float*)d_in[1];
  const float* bq = (const float*)d_in[2];
  const float* Wv = (const float*)d_in[3];
  const float* bv = (const float*)d_in[4];
  const float* Wo = (const float*)d_in[5];
  const float* bo = (const float*)d_in[6];
  float* out = (float*)d_out;

  char* ws = (char*)d_ws;
  bf16* q_s     = (bf16*)(ws);                        // 8 MB   [bh][n][d]
  bf16* vT_s    = (bf16*)(ws + ((size_t)8 << 20));    // 8 MB   [bh][d][n]
  bf16* outflat = (bf16*)(ws + ((size_t)16 << 20));   // 8 MB   [b*n][dim]
  float* sq_s   = (float*)(ws + ((size_t)24 << 20));  // 256 KB [bh][n]
  bf16* x_bf    = (bf16*)(ws + ((size_t)25 << 20));   // 8 MB   [b*n][dim]
  bf16* wq_bf   = (bf16*)(ws + ((size_t)33 << 20));   // 2 MB
  bf16* wv_bf   = (bf16*)(ws + ((size_t)35 << 20));   // 2 MB
  bf16* wo_bf   = (bf16*)(ws + ((size_t)37 << 20));   // 2 MB

  dim3 blk(256);
  k_cvt_bf16<<<1024, blk, 0, stream>>>(x,  x_bf,  TOK * DIM);
  k_cvt_bf16<<<512,  blk, 0, stream>>>(Wq, wq_bf, DIM * DIM);
  k_cvt_bf16<<<512,  blk, 0, stream>>>(Wv, wv_bf, DIM * DIM);
  k_cvt_bf16<<<512,  blk, 0, stream>>>(Wo, wo_bf, DIM * DIM);

  k_proj_q<<<128, blk, 0, stream>>>(x_bf, wq_bf, bq, q_s, sq_s);
  k_proj_v<<<128, blk, 0, stream>>>(x_bf, wv_bf, bv, vT_s);
  k_attn  <<<512, blk, 0, stream>>>(q_s, vT_s, sq_s, outflat);
  k_proj_o<<<128, blk, 0, stream>>>(outflat, wo_bf, bo, out);
}
